// ChernClassCalculator_20057497272550
// MI455X (gfx1250) — matrix operand order repaired and checked by reading, not compiled, not run on hardware
//
#include <hip/hip_runtime.h>
#include <math.h>

// ---------------------------------------------------------------------------
// ChernClassCalculator for MI455X (gfx1250, wave32, WMMA bf16)
//
// A = W + D, D = 0.1*diag(x @ Wc).  Algebra: A^2 - A^T A = S + Wasym*D with
//   S = W^2 - W^T W (sample-independent), Wasym = W - W^T.
// So F = S + Wasym*diag(d) + 0.01*A^3; per sample only the A^2 and
// A^3 = A^2*A WMMA chains remain.  A^T lives in LDS (B-operands contiguous);
// A-operand rows for phase 2 are rebuilt per-strip row-major straight from
// global W (coalesced) so NO LDS gathers remain anywhere; all fragment loads
// are explicit 128-bit (ds_load_b128).
// ---------------------------------------------------------------------------

typedef __attribute__((ext_vector_type(16))) __bf16 v16bf;
typedef __attribute__((ext_vector_type(8)))  float  v8f;
typedef __attribute__((ext_vector_type(4)))  unsigned int v4u;

#define DMODEL   256
#define LDA      264        // bf16 row stride: 528 B rows, 16B-aligned, +4 bank stagger
#define NTHREADS 256
#define NWAVES   8
#define STG      33         // f32 stage row stride (32x32 block + pad)

#define SMEM_BYTES ((2 * DMODEL * LDA + 32 * LDA) * 2 /* sAT + sA2 + strip */ \
                    + (NWAVES * 32 * STG + DMODEL + DMODEL + 2 * NTHREADS) * 4)

union FAcc { v8f v; float e[8]; };

__device__ __forceinline__ unsigned short f32_to_bf16(float f) {
  unsigned int u = __float_as_uint(f);
  u += 0x7FFFu + ((u >> 16) & 1u);          // round to nearest even
  return (unsigned short)(u >> 16);
}

__device__ __forceinline__ v8f wmma_bf16(v16bf a, v16bf b, v8f c) {
  return __builtin_amdgcn_wmma_f32_16x16x32_bf16(false, a, false, b,
                                                 (short)0, c, false, false);
}

// A-operand fragment (rows r0..r0+15, K=k0..k0+31) from row-major bf16 LDS.
// ISA 16-bit A layout: lanes 0-15 K 0..7 / 16..23, lanes 16-31 K 8..15 / 24..31.
// Two aligned 16B runs per lane -> 2x ds_load_b128.
__device__ __forceinline__ v16bf load_arow(const unsigned short* S, int r0,
                                           int k0, int lane) {
  const unsigned short* p = S + (r0 + (lane & 15)) * LDA + k0 + ((lane >> 4) << 3);
  union { v16bf v; v4u q[2]; } f;
  f.q[0] = *(const v4u*)p;          // K chunk 0..7 (or 8..15)
  f.q[1] = *(const v4u*)(p + 16);   // K chunk 16..23 (or 24..31)
  return f.v;
}

// B-operand fragment (cols c0..c0+15 of M, K=k0..k0+31) when LDS holds M^T:
// lane l holds column N=l&15, K=(l>>4)*8+0..7 and 16+(l>>4)*8+0..7 -> two aligned 16B runs.
__device__ __forceinline__ v16bf load_bcol_from_T(const unsigned short* T, int k0,
                                                  int c0, int lane) {
  const unsigned short* p = T + (c0 + (lane & 15)) * LDA + k0 + ((lane >> 4) << 3);
  union { v16bf v; v4u q[2]; } f;
  f.q[0] = *(const v4u*)p;
  f.q[1] = *(const v4u*)(p + 16);
  return f.v;
}

__device__ __forceinline__ void store_tile_bf16(unsigned short* Dst, v8f acc,
                                                int Ti, int Tj, int lane) {
  FAcc t; t.v = acc;
  const int rbase = Ti * 16 + (lane >> 4) * 8;
  const int cb    = Tj * 16 + (lane & 15);
#pragma unroll
  for (int v = 0; v < 8; ++v) Dst[(rbase + v) * LDA + cb] = f32_to_bf16(t.e[v]);
}

// Phase 2: one 2x2 block of A^2 tiles; A rows from the row-major strip buffer
// (strip-local rows 0..31), A cols from sAT.  All loads contiguous.
__device__ __forceinline__ void a2_strip_block(const unsigned short* sStrip,
                                               const unsigned short* sAT,
                                               unsigned short* sA2,
                                               int Rb /*global tile-row*/,
                                               int Cb, int lane) {
  v8f a00 = {}, a01 = {}, a10 = {}, a11 = {};
  for (int kc = 0; kc < 8; ++kc) {
    v16bf r0 = load_arow(sStrip, 0, kc * 32, lane);    // strip-local rows 0..15
    v16bf r1 = load_arow(sStrip, 16, kc * 32, lane);   // strip-local rows 16..31
    v16bf b0 = load_bcol_from_T(sAT, kc * 32, Cb * 16, lane);
    v16bf b1 = load_bcol_from_T(sAT, kc * 32, (Cb + 1) * 16, lane);
    a00 = wmma_bf16(r0, b0, a00);
    a01 = wmma_bf16(r0, b1, a01);
    a10 = wmma_bf16(r1, b0, a10);
    a11 = wmma_bf16(r1, b1, a11);
  }
  store_tile_bf16(sA2, a00, Rb, Cb, lane);
  store_tile_bf16(sA2, a01, Rb, Cb + 1, lane);
  store_tile_bf16(sA2, a10, Rb + 1, Cb, lane);
  store_tile_bf16(sA2, a11, Rb + 1, Cb + 1, lane);
}

// Phase 3: 2x2 block of F tiles, F = S + Wasym*diag(d) + 0.01*(A^2 * A).
__device__ __forceinline__ void f_block(const unsigned short* sAT,
                                        const unsigned short* sA2,
                                        const float* __restrict__ Sg,
                                        const float* __restrict__ Wag,
                                        const float* sdiag,
                                        int Rb, int Cb, int lane, FAcc out[2][2]) {
  v8f a00 = {}, a01 = {}, a10 = {}, a11 = {};
  for (int kc = 0; kc < 8; ++kc) {
    v16bf r0 = load_arow(sA2, Rb * 16, kc * 32, lane);
    v16bf r1 = load_arow(sA2, (Rb + 1) * 16, kc * 32, lane);
    v16bf b0 = load_bcol_from_T(sAT, kc * 32, Cb * 16, lane);
    v16bf b1 = load_bcol_from_T(sAT, kc * 32, (Cb + 1) * 16, lane);
    a00 = wmma_bf16(r0, b0, a00);
    a01 = wmma_bf16(r0, b1, a01);
    a10 = wmma_bf16(r1, b0, a10);
    a11 = wmma_bf16(r1, b1, a11);
  }
  v8f acc[2][2] = {{a00, a01}, {a10, a11}};
  const int rr = (lane >> 4) * 8;
  const int cc = lane & 15;
#pragma unroll
  for (int ti = 0; ti < 2; ++ti) {
#pragma unroll
    for (int tj = 0; tj < 2; ++tj) {
      const int gr0 = (Rb + ti) * 16 + rr;
      const int gc  = (Cb + tj) * 16 + cc;
      const float dj = sdiag[gc];
      FAcc a; a.v = acc[ti][tj];
#pragma unroll
      for (int v = 0; v < 8; ++v) {
        const int idx = (gr0 + v) * DMODEL + gc;
        out[ti][tj].e[v] = Sg[idx] + Wag[idx] * dj + 0.01f * a.e[v];
      }
    }
  }
}

__device__ __forceinline__ void stage_block(float* st, const FAcc B[2][2], int lane) {
  const int rr = (lane >> 4) * 8;
  const int cc = lane & 15;
#pragma unroll
  for (int ti = 0; ti < 2; ++ti)
#pragma unroll
    for (int tj = 0; tj < 2; ++tj)
#pragma unroll
      for (int v = 0; v < 8; ++v)
        st[(ti * 16 + rr + v) * STG + tj * 16 + cc] = B[ti][tj].e[v];
}

// ---- one-time precompute: S = W^2 - W^T W, Wasym = W - W^T (into d_ws) ----
__global__ __launch_bounds__(NTHREADS)
void chern_precompute_kernel(const float* __restrict__ W,
                             float* __restrict__ S, float* __restrict__ Wa) {
  const int i = blockIdx.x, j = threadIdx.x;
  float s1 = 0.f, s2 = 0.f;
  for (int k = 0; k < DMODEL; ++k) {
    const float wkj = W[k * DMODEL + j];
    s1 += W[i * DMODEL + k] * wkj;
    s2 += W[k * DMODEL + i] * wkj;
  }
  S[i * DMODEL + j]  = s1 - s2;
  Wa[i * DMODEL + j] = W[i * DMODEL + j] - W[j * DMODEL + i];
}

__global__ __launch_bounds__(NTHREADS)
void chern_class_kernel(const float* __restrict__ x,
                        const float* __restrict__ Wconn,
                        const float* __restrict__ Wcurv,
                        const float* __restrict__ Sg,
                        const float* __restrict__ Wag,
                        float* __restrict__ out, int M) {
  extern __shared__ char smem_raw[];
  unsigned short* sAT    = (unsigned short*)smem_raw;   // A^T, bf16, row-major
  unsigned short* sA2    = sAT + DMODEL * LDA;          // A^2, bf16, row-major
  unsigned short* sStrip = sA2 + DMODEL * LDA;          // 32-row strip of A
  float* stage = (float*)(sStrip + 32 * LDA);           // NWAVES x 32 x STG
  float* sx    = stage + NWAVES * 32 * STG;
  float* sdiag = sx + DMODEL;
  float* red   = sdiag + DMODEL;                        // 2 * NTHREADS

  const int tid  = threadIdx.x;
  const int lane = tid & 31;
  const int wave = tid >> 5;
  const int m    = blockIdx.x;

  // ---- Phase 1: d = 0.1*(x_m @ Wcurv); build A^T (bf16) in LDS ----
  sx[tid] = x[(size_t)m * DMODEL + tid];
  __syncthreads();
  float e = 0.f;
  for (int k = 0; k < DMODEL; ++k) e += sx[k] * Wcurv[k * DMODEL + tid];
  sdiag[tid] = 0.1f * e;                 // only sdiag[tid] read before barrier
  for (int j = 0; j < DMODEL; ++j) {     // A[j][tid] -> AT[tid][j]; coalesced read
    float a = Wconn[j * DMODEL + tid];
    if (j == tid) a += sdiag[tid];
    sAT[tid * LDA + j] = f32_to_bf16(a);
  }
  __syncthreads();

  // ---- Phase 2: A^2 = A*A, strip-structured (zero LDS gathers) ----
  for (int s = 0; s < 16; ++s) {
    // cooperative rebuild of rows 32s..32s+31 of A, row-major, from global W
    for (int r = 0; r < 32; ++r) {
      const int gr = 32 * s + r;
      float a = Wconn[gr * DMODEL + tid];
      if (gr == tid) a += sdiag[tid];
      sStrip[r * LDA + tid] = f32_to_bf16(a);
    }
    if (s < 15)  // pull next W strip toward the WGP (global_prefetch_b8)
      __builtin_prefetch(&Wconn[(32 * (s + 1)) * DMODEL + tid], 0, 1);
    __syncthreads();
    a2_strip_block(sStrip, sAT, sA2, 2 * s, 2 * wave, lane);
    __syncthreads();
  }

  // ---- Phase 3: symmetric 2x2 block-pairs; tr(F), tr(F*F) on the fly ----
  float trF = 0.f, trF2 = 0.f;
  float* st = stage + wave * 32 * STG;
  const int rr = (lane >> 4) * 8;
  const int cc = lane & 15;
  int bp = 0;
  for (int Ib = 0; Ib < 8; ++Ib) {
    for (int Jb = Ib; Jb < 8; ++Jb, ++bp) {
      if ((bp % NWAVES) != wave) continue;          // wave-uniform; EXEC stays full
      FAcc B1[2][2];
      if (Ib == Jb) {
        f_block(sAT, sA2, Sg, Wag, sdiag, 2 * Ib, 2 * Ib, lane, B1);
        stage_block(st, B1, lane);
#pragma unroll
        for (int ti = 0; ti < 2; ++ti)
#pragma unroll
          for (int tj = 0; tj < 2; ++tj)
#pragma unroll
            for (int v = 0; v < 8; ++v) {
              const int a = ti * 16 + rr + v, b = tj * 16 + cc;
              trF2 += B1[ti][tj].e[v] * st[b * STG + a];     // F_ij * F_ji
              if (a == b) trF += B1[ti][tj].e[v];
            }
      } else {
        FAcc B2[2][2];
        f_block(sAT, sA2, Sg, Wag, sdiag, 2 * Jb, 2 * Ib, lane, B2);  // mirror
        stage_block(st, B2, lane);
        f_block(sAT, sA2, Sg, Wag, sdiag, 2 * Ib, 2 * Jb, lane, B1);
#pragma unroll
        for (int ti = 0; ti < 2; ++ti)
#pragma unroll
          for (int tj = 0; tj < 2; ++tj)
#pragma unroll
            for (int v = 0; v < 8; ++v) {
              const int a = ti * 16 + rr + v, b = tj * 16 + cc;
              trF2 += 2.f * B1[ti][tj].e[v] * st[b * STG + a];
            }
      }
    }
  }

  // ---- Block reduction + outputs ----
  red[tid] = trF;
  red[NTHREADS + tid] = trF2;
  __syncthreads();
  for (int s2 = NTHREADS / 2; s2 > 0; s2 >>= 1) {
    if (tid < s2) {
      red[tid] += red[tid + s2];
      red[NTHREADS + tid] += red[NTHREADS + tid + s2];
    }
    __syncthreads();
  }
  if (tid == 0) {
    const float tF  = red[0];
    const float tF2 = red[NTHREADS];
    const float PI  = 3.14159265358979323846f;
    const float c1  = tF / (2.f * PI);
    const float c2  = (tF2 - tF * tF) / (8.f * PI * PI);
    const float c2d = c2 / (fabsf(c1) + 1e-8f);
    out[m]         = c1;     // tuple order: c1, c2, c2/|c1|, trace_F
    out[M + m]     = c2;
    out[2 * M + m] = c2d;
    out[3 * M + m] = tF;
  }
}

extern "C" void kernel_launch(void* const* d_in, const int* in_sizes, int n_in,
                              void* d_out, int out_size, void* d_ws, size_t ws_size,
                              hipStream_t stream) {
  const float* x  = (const float*)d_in[0];  // [B, N, 256] f32
  const float* Wc = (const float*)d_in[1];  // connection_form [256,256]
  const float* Wk = (const float*)d_in[2];  // curvature_weight [256,256]
  float* out = (float*)d_out;               // 4*M floats (c1, c2, c2/|c1|, trF)
  const int M = in_sizes[0] / DMODEL;       // 1024 samples

  // d_ws: S (256KB f32) + Wasym (256KB f32); written then read every call.
  float* Sg  = (float*)d_ws;
  float* Wag = Sg + DMODEL * DMODEL;

  (void)n_in; (void)out_size; (void)ws_size;
  hipFuncSetAttribute((const void*)chern_class_kernel,
                      hipFuncAttributeMaxDynamicSharedMemorySize,
                      (int)SMEM_BYTES);
  chern_precompute_kernel<<<DMODEL, NTHREADS, 0, stream>>>(Wc, Sg, Wag);
  chern_class_kernel<<<M, NTHREADS, SMEM_BYTES, stream>>>(x, Wc, Wk, Sg, Wag,
                                                          out, M);
}
